// SelfAttention_2843268350308
// MI455X (gfx1250) — matrix operand order repaired and checked by reading, not compiled, not run on hardware
//
#include <hip/hip_runtime.h>
#include <hip/hip_bf16.h>

// ---------- types ----------
typedef _Float16 half16 __attribute__((ext_vector_type(16)));
typedef _Float16 half8  __attribute__((ext_vector_type(8)));
typedef float    float8 __attribute__((ext_vector_type(8)));
typedef float    vfloat4 __attribute__((ext_vector_type(4)));

// ---------- problem constants ----------
#define EMBED 512
#define DK    64
#define BATCH 4
#define SEQ   4096
#define NTILES (BATCH * SEQ / 16)      // 1024 row-tiles of 16

// workspace layout (in _Float16 element units)
//  [0      , 32768 )  Wq fragments: [16 ksteps][4 ntiles][32 lanes][16 halves]
//  [32768  , 65536 )  Wk fragments: same
//  [65536  , +1M   )  Q fragments : [1024 tiles][2 kfrag][32 lanes][16 halves]
//  [65536+1M, +1M  )  K fragments : same
#define WF_HALVES   32768
#define QF_BASE     65536
#define FRAG_HALVES (NTILES * 2 * 32 * 16)   // 1,048,576
#define KF_BASE     (QF_BASE + FRAG_HALVES)

// =====================================================================
// Kernel 1: swizzle Wq / Wk (f32 row-major [512][64]) into f16 WMMA
// B-fragments. B-matrix layout (16x16x32 f16): lane n holds column n&15,
// hi = n>>4; slot j -> k = (j&7) + 8*hi + 16*(j>>3).
// =====================================================================
__global__ void wswizzle_kernel(const float* __restrict__ Wq,
                                const float* __restrict__ Wk,
                                _Float16* __restrict__ ws) {
    const int m = blockIdx.x;                        // 0 = Wq, 1 = Wk
    const float* __restrict__ W = m ? Wk : Wq;
    _Float16* __restrict__ out = ws + m * WF_HALVES;
    for (int h = threadIdx.x; h < WF_HALVES; h += blockDim.x) {
        int j    = h & 15;
        int lane = (h >> 4) & 31;
        int nt   = (h >> 9) & 3;
        int ks   = h >> 11;
        int hi   = lane >> 4;
        int col  = nt * 16 + (lane & 15);
        int k    = ks * 32 + (j & 7) + 8 * hi + 16 * (j >> 3);
        out[h] = (_Float16)W[k * DK + col];
    }
}

// =====================================================================
// Kernel 2: projection. One wave per 16-row tile of Q or K.
//   P[16x64] = X[16x512] @ W[512x64] + bias  via v_wmma_f32_16x16x32_f16.
// Q tiles additionally absorb the 1/sqrt(d_k)=0.125 score scale.
// Result is staged in LDS and re-emitted as pre-swizzled WMMA fragments:
//   Q tiles -> A-matrix layout: slot j -> k = (j&7) + 8*hi + 16*(j>>3)
//   K tiles -> B-matrix layout: slot j -> k = (j&7) + 8*hi + 16*(j>>3)
// =====================================================================
__global__ void proj_kernel(const float* __restrict__ q_in,
                            const float* __restrict__ k_in,
                            const float* __restrict__ bq,
                            const float* __restrict__ bk,
                            _Float16* __restrict__ ws) {
    __shared__ __align__(16) _Float16 sm[8][16 * DK];   // 2 KB per wave

    const int warp = threadIdx.x >> 5;
    const int lane = threadIdx.x & 31;
    const int g    = blockIdx.x * 8 + warp;             // 0..2047
    const int mat  = g >> 10;                           // 0 = Q, 1 = K
    const int t    = g & (NTILES - 1);                  // row-tile index

    const float* __restrict__ X    = mat ? k_in : q_in;
    const float* __restrict__ bias = mat ? bk : bq;
    const float scale = mat ? 1.0f : 0.125f;            // fold 1/sqrt(64) into Q
    const _Float16* __restrict__ wf = ws + mat * WF_HALVES;
    _Float16* __restrict__ outf = ws + QF_BASE + mat * FRAG_HALVES
                                     + (size_t)t * 1024;

    const int r15 = lane & 15;
    const int hi  = lane >> 4;
    const float* __restrict__ Xrow = X + (size_t)(t * 16 + r15) * EMBED;

    float8 acc[4] = {};                                 // 4 n-tiles of 16

    for (int ks = 0; ks < 16; ++ks) {
        const int base = ks * 32;
        // A-fragment gather: lane holds row r15; slots 0..7 -> k = base+8*hi+u,
        // slots 8..15 -> k = base+16+8*hi+u  (two contiguous 32B chunks).
        // Inputs are read exactly once -> non-temporal (don't pollute L2).
        vfloat4 f[4];
        f[0] = __builtin_nontemporal_load((const vfloat4*)(Xrow + base + 8 * hi));
        f[1] = __builtin_nontemporal_load((const vfloat4*)(Xrow + base + 8 * hi + 4));
        f[2] = __builtin_nontemporal_load((const vfloat4*)(Xrow + base + 16 + 8 * hi));
        f[3] = __builtin_nontemporal_load((const vfloat4*)(Xrow + base + 16 + 8 * hi + 4));
        half16 a;
#pragma unroll
        for (int u = 0; u < 16; ++u) a[u] = (_Float16)f[u >> 2][u & 3];

#pragma unroll
        for (int nt = 0; nt < 4; ++nt) {
            half16 b = *(const half16*)(wf + ((ks * 4 + nt) * 32 + lane) * 16);
            acc[nt] = __builtin_amdgcn_wmma_f32_16x16x32_f16(
                false, a, false, b, (short)0, acc[nt], false, false);
        }
    }

    // bias (+scale) + dump C (lane l, vgpr r -> row r+8*hi, col nt*16+(l&15))
#pragma unroll
    for (int nt = 0; nt < 4; ++nt) {
        float bv = bias[nt * 16 + r15];
#pragma unroll
        for (int r = 0; r < 8; ++r)
            sm[warp][(r + 8 * hi) * DK + nt * 16 + r15] =
                (_Float16)((acc[nt][r] + bv) * scale);
    }
    __syncthreads();

    // re-emit as fragments: per lane, 4 chunks of 8 contiguous halves (16 B)
#pragma unroll
    for (int kf = 0; kf < 2; ++kf)
#pragma unroll
        for (int c = 0; c < 2; ++c) {
            int col0 = mat ? (kf * 32 + 8 * hi + 16 * c)      // K: B-layout
                           : (kf * 32 + 8 * hi + 16 * c);     // Q: A-layout
            half8 v = *(const half8*)&sm[warp][r15 * DK + col0];
            *(half8*)(outf + (kf * 32 + lane) * 16 + c * 8) = v;
        }
}

// =====================================================================
// Kernel 3: scores = (Q/8) K^T. One wave per 16x128 strip.
// Fragments are pre-swizzled -> pure coalesced b128 loads + 2 WMMAs/tile.
// Output (256 MB > 192 MB L2) is streamed with non-temporal stores so the
// 4.25 MB fragment working set stays L2-resident; HBM sees only the output.
// =====================================================================
__global__ void score_kernel(const _Float16* __restrict__ ws,
                             float* __restrict__ out) {
    const int warp = threadIdx.x >> 5;
    const int lane = threadIdx.x & 31;
    const int g    = blockIdx.x * 8 + warp;     // 0..32767
    const int nb   = g & 31;                    // 128-col block
    const int rem  = g >> 5;
    const int mt   = rem & 255;                 // 16-row tile in batch
    const int b    = rem >> 8;                  // batch

    const int r15 = lane & 15;
    const int hi  = lane >> 4;

    const _Float16* __restrict__ qf =
        ws + QF_BASE + (size_t)(b * 256 + mt) * 1024;
    half16 a0 = *(const half16*)(qf + lane * 16);
    half16 a1 = *(const half16*)(qf + (32 + lane) * 16);

    float* __restrict__ outp =
        out + ((size_t)(b * SEQ + mt * 16)) * SEQ + nb * 128;

#pragma unroll
    for (int nt = 0; nt < 8; ++nt) {
        const int ktile = b * 256 + nb * 8 + nt;
        const _Float16* __restrict__ kf =
            ws + KF_BASE + (size_t)ktile * 1024;
        half16 b0 = *(const half16*)(kf + lane * 16);
        half16 b1 = *(const half16*)(kf + (32 + lane) * 16);

        float8 acc = {};
        acc = __builtin_amdgcn_wmma_f32_16x16x32_f16(
            false, a0, false, b0, (short)0, acc, false, false);
        acc = __builtin_amdgcn_wmma_f32_16x16x32_f16(
            false, a1, false, b1, (short)0, acc, false, false);

        // C layout: lane l, vgpr r -> row r+8*hi, col (l&15); NT stream out
#pragma unroll
        for (int r = 0; r < 8; ++r)
            __builtin_nontemporal_store(
                acc[r], &outp[(size_t)(r + 8 * hi) * SEQ + nt * 16 + r15]);
    }
}

// =====================================================================
// launch
// =====================================================================
extern "C" void kernel_launch(void* const* d_in, const int* in_sizes, int n_in,
                              void* d_out, int out_size, void* d_ws, size_t ws_size,
                              hipStream_t stream) {
    const float* query_in = (const float*)d_in[0];
    const float* key_in   = (const float*)d_in[1];
    // d_in[2] = value_in (unused by reference output)
    const float* Wq = (const float*)d_in[3];
    const float* bq = (const float*)d_in[4];
    const float* Wk = (const float*)d_in[5];
    const float* bk = (const float*)d_in[6];
    // d_in[7], d_in[8] = Wv, bv (unused)
    float* out = (float*)d_out;
    _Float16* ws = (_Float16*)d_ws;

    // 1) swizzle weights into WMMA B-fragments (128 KB, L2-resident)
    wswizzle_kernel<<<2, 256, 0, stream>>>(Wq, Wk, ws);

    // 2) Q/K projections -> pre-swizzled f16 fragments (2048 waves)
    proj_kernel<<<256, 256, 0, stream>>>(query_in, key_in, bq, bk, ws);

    // 3) scores (32768 waves, store-bandwidth bound)
    score_kernel<<<4096, 256, 0, stream>>>(ws, out);
}
